// LRUModel_49340584297000
// MI455X (gfx1250) — matrix operand order repaired and checked by reading, not compiled, not run on hardware
//
#include <hip/hip_runtime.h>
#include <math.h>

// Problem constants (match reference)
#define DD   256
#define SS   2048
#define BB   8
#define VV   6
#define ROWS (BB * SS)          // 16384 token rows
#define MT   32                 // GEMM M-tile per workgroup
#define LDA  264                // padded LDS row stride in halves (bank spread)
#define WSLOT 65536             // one 256x256 f16 matrix

typedef __attribute__((ext_vector_type(16))) _Float16 v16h;
typedef __attribute__((ext_vector_type(8)))  _Float16 v8h;
typedef __attribute__((ext_vector_type(8)))  float    v8f;
typedef __attribute__((ext_vector_type(4)))  unsigned int u32x4;
typedef __attribute__((ext_vector_type(8)))  int      i32x8;
typedef __attribute__((ext_vector_type(4)))  int      i32x4;

// Tensor Data Mover availability (guarded: falls back to direct loads)
#if defined(__gfx1250__) && __has_builtin(__builtin_amdgcn_tensor_load_to_lds)
#define HAVE_TDM 1
#else
#define HAVE_TDM 0
#endif

static __device__ __forceinline__ v16h cat8(v8h lo, v8h hi) {
  return __builtin_shufflevector(lo, hi, 0,1,2,3,4,5,6,7,8,9,10,11,12,13,14,15);
}

// ---------------------------------------------------------------------------
// Convert + transpose a 256x256 f32 weight W[k][n] into WT16[n][k] halves so
// B-fragments become contiguous 32B loads per lane.
__global__ void __launch_bounds__(256)
convw_k(const float* __restrict__ W, _Float16* __restrict__ WT) {
  const int n = blockIdx.x;
  const int k = threadIdx.x;
  WT[(size_t)n * DD + k] = (_Float16)W[(size_t)k * DD + n];
}

// ---------------------------------------------------------------------------
// U[row, d] = Wi0[x[row], d] + bi0[d]   (one-hot @ Wi0 is a row gather)
__global__ void __launch_bounds__(256)
embed_k(const int* __restrict__ x, const float* __restrict__ Wi0,
        const float* __restrict__ bi0, float* __restrict__ U) {
  const int idx = blockIdx.x * 256 + threadIdx.x;   // over ROWS*DD
  const int row = idx >> 8;
  const int d   = idx & 255;
  const int tok = x[row];
  U[idx] = Wi0[tok * DD + d] + bi0[d];
}

// ---------------------------------------------------------------------------
// WMMA GEMM: C[ROWS,256] = A[ROWS,256](f32) @ W (given as WT16[n][k]) + bias
// grid = (ROWS/MT, 4), block = 128 (4 waves); each wave: two 16x16 M-tiles of
// one 16-wide N-slice (register blocking: each B-fragment feeds 2 WMMAs).
// A-tile staged into LDS via the Tensor Data Mover when available.
__global__ void __launch_bounds__(128)
wmma_gemm_k(const float* __restrict__ A, const _Float16* __restrict__ WT,
            const float* __restrict__ bias, float* __restrict__ C) {
  __shared__ __align__(16) _Float16 Alds[MT * LDA];
  const int m0 = blockIdx.x * MT;

#if HAVE_TDM
  __shared__ __align__(16) float Astage[MT * DD];
  if (threadIdx.x < 32) {            // wave 0 issues one TDM descriptor
    const unsigned long long ga =
        (unsigned long long)(const void*)(A + (size_t)m0 * DD);
    const unsigned int lds = (unsigned int)(unsigned long long)(void*)&Astage[0];
    u32x4 g0;
    g0[0] = 1u;                                        // count=1, user mode
    g0[1] = lds;                                       // lds_addr (bytes)
    g0[2] = (unsigned int)ga;                          // global_addr[31:0]
    g0[3] = (unsigned int)((ga >> 32) & 0x01FFFFFFu)   // global_addr[56:32]
            | (2u << 30);                              // type=2 ("image")
    i32x8 g1;
    g1[0] = (int)(2u << 16);          // workgroup_mask=0, data_size=4B
    g1[1] = (int)(DD << 16);          // tensor_dim0[15:0]=256 at bits[63:48]
    g1[2] = (int)(ROWS << 16);        // dim0 hi=0 | tensor_dim1[15:0]=16384
    g1[3] = (int)(DD << 16);          // dim1 hi=0 | tile_dim0=256
    g1[4] = MT;                       // tile_dim1=32, tile_dim2=0
    g1[5] = DD;                       // tensor_dim0_stride=256
    g1[6] = 0;
    g1[7] = 0;
    const i32x4 z4 = {0, 0, 0, 0};
#if __clang_major__ >= 23
    const i32x8 z8 = {0, 0, 0, 0, 0, 0, 0, 0};
    __builtin_amdgcn_tensor_load_to_lds(g0, g1, z4, z4, z8, 0);
#else
    __builtin_amdgcn_tensor_load_to_lds(g0, g1, z4, z4, 0);
#endif
#if __has_builtin(__builtin_amdgcn_s_wait_tensorcnt)
    __builtin_amdgcn_s_wait_tensorcnt(0);
#endif
  }
  __syncthreads();
  for (int i = threadIdx.x; i < MT * DD; i += 128)
    Alds[(i >> 8) * LDA + (i & 255)] = (_Float16)Astage[i];
  __syncthreads();
#else
  for (int i = threadIdx.x; i < MT * DD; i += 128) {
    const int r = i >> 8, k = i & 255;
    Alds[r * LDA + k] = (_Float16)A[(size_t)(m0 + r) * DD + k];
  }
  __syncthreads();
#endif

  const int wave = threadIdx.x >> 5;
  const int lane = threadIdx.x & 31;
  const int col  = blockIdx.y * 64 + wave * 16 + (lane & 15);
  const int hi   = (lane & 16) ? 1 : 0;   // upper half-wave: K+16..K+31 slice
  const int mA   = lane & 15;

  v8f c0, c1;
  const float bv = bias[col];
#pragma unroll
  for (int r = 0; r < 8; ++r) { c0[r] = bv; c1[r] = bv; }

#pragma unroll
  for (int ks = 0; ks < 8; ++ks) {
    const int kb = ks * 32;
    v8h a0lo = *(const v8h*)&Alds[mA * LDA + kb + hi * 16];
    v8h a0hi = *(const v8h*)&Alds[mA * LDA + kb + hi * 16 + 8];
    v8h a1lo = *(const v8h*)&Alds[(16 + mA) * LDA + kb + hi * 16];
    v8h a1hi = *(const v8h*)&Alds[(16 + mA) * LDA + kb + hi * 16 + 8];
    v16h b = *(const v16h*)&WT[(size_t)col * DD + kb + hi * 16];
    c0 = __builtin_amdgcn_wmma_f32_16x16x32_f16(false, cat8(a0lo, a0hi), false,
                                                b, (short)0, c0, false, false);
    c1 = __builtin_amdgcn_wmma_f32_16x16x32_f16(false, cat8(a1lo, a1hi), false,
                                                b, (short)0, c1, false, false);
  }

#pragma unroll
  for (int r = 0; r < 8; ++r) {
    const int m = m0 + (hi ? r + 8 : r);
    C[(size_t)m * DD + col]        = c0[r];
    C[(size_t)(m + 16) * DD + col] = c1[r];
  }
}

// ---------------------------------------------------------------------------
// Linear RNN scan: h_t = h_{t-1} @ Wh + u_t for t in [0,2048), batch rows in
// WMMA M dim. Single workgroup, 16 waves; wave w owns columns [16w,16w+16).
// Wh slice lives in VGPRs (8 B-fragments), state h (8x256) in LDS.
__global__ void __launch_bounds__(512)
wmma_scan_k(const float* __restrict__ U, const _Float16* __restrict__ WhT,
            float* __restrict__ H) {
  __shared__ __align__(16) _Float16 h16[16 * LDA];
  for (int i = threadIdx.x; i < 16 * LDA; i += 512) h16[i] = (_Float16)0.f;
  __syncthreads();

  const int wave = threadIdx.x >> 5;
  const int lane = threadIdx.x & 31;
  const int col  = wave * 16 + (lane & 15);
  const int hi   = (lane & 16) ? 1 : 0;
  const int mA   = lane & 15;

  // Preload this wave's Wh column slice as 8 B-fragments (stays in VGPRs).
  v16h bfrag[8];
#pragma unroll
  for (int ks = 0; ks < 8; ++ks)
    bfrag[ks] = *(const v16h*)&WhT[(size_t)col * DD + ks * 32 + hi * 16];

  for (int t = 0; t < SS; ++t) {
    // c := u_t  (rows 0..7 = batch; rows 8..15 stay zero)
    v8f c;
#pragma unroll
    for (int r = 0; r < 8; ++r)
      c[r] = hi ? 0.f : U[((size_t)r * SS + t) * DD + col];

    if (t + 1 < SS)
      __builtin_prefetch(&U[((size_t)(mA & 7) * SS + (t + 1)) * DD + col], 0, 0);

    // c += h_{t-1} @ Wh (K=256 as 8 WMMA k-steps)
#pragma unroll
    for (int ks = 0; ks < 8; ++ks) {
      v8h alo = *(const v8h*)&h16[mA * LDA + ks * 32 + hi * 16];
      v8h ahi = *(const v8h*)&h16[mA * LDA + ks * 32 + hi * 16 + 8];
      c = __builtin_amdgcn_wmma_f32_16x16x32_f16(false, cat8(alo, ahi), false,
                                                 bfrag[ks], (short)0, c,
                                                 false, false);
    }

    __syncthreads();          // all waves done reading previous h
    if (!hi) {
#pragma unroll
      for (int r = 0; r < 8; ++r) {
        H[((size_t)r * SS + t) * DD + col] = c[r];
        h16[r * LDA + col] = (_Float16)c[r];
      }
    }
    __syncthreads();          // new h visible to all waves
  }
}

// ---------------------------------------------------------------------------
__global__ void zero_k(float* __restrict__ p, int n) {
  const int i = blockIdx.x * blockDim.x + threadIdx.x;
  if (i < n) p[i] = 0.f;
}

// Per-feature sum / sumsq over 16384 rows (coalesced, f32 atomics to combine)
__global__ void __launch_bounds__(256)
stats_k(const float* __restrict__ Y, float* __restrict__ stats) {
  const int d = threadIdx.x;
  const int row0 = blockIdx.x * 512;
  float s = 0.f, s2 = 0.f;
  for (int r = 0; r < 512; ++r) {
    const float v = Y[(size_t)(row0 + r) * DD + d];
    s += v; s2 += v * v;
  }
  atomicAdd(&stats[d], s);
  atomicAdd(&stats[DD + d], s2);
}

// BatchNorm + GELU(tanh) + residual, in place on H
__global__ void __launch_bounds__(256)
norm_gelu_res_k(const float* __restrict__ Y, const float* __restrict__ stats,
                const float* __restrict__ scale, const float* __restrict__ shift,
                float* __restrict__ Hio) {
  const int idx = blockIdx.x * 256 + threadIdx.x;   // over ROWS*DD
  const int d = idx & 255;
  const float invN = 1.f / (float)ROWS;
  const float mu  = stats[d] * invN;
  const float var = stats[DD + d] * invN - mu * mu;
  const float y   = Y[idx];
  const float yh  = (y - mu) * rsqrtf(var + 1e-5f) * scale[d] + shift[d];
  const float g   = 0.5f * yh *
      (1.f + tanhf(0.7978845608028654f * (yh + 0.044715f * yh * yh * yh)));
  Hio[idx] = g + Hio[idx];
}

// Readout: out[row, v] = dot(H[row,:], Wr[:,v]) + br[v]   (N=6, VALU kernel)
__global__ void __launch_bounds__(256)
readout_k(const float* __restrict__ H, const float* __restrict__ Wr,
          const float* __restrict__ br, float* __restrict__ out) {
  const int idx = blockIdx.x * blockDim.x + threadIdx.x;
  if (idx >= ROWS * VV) return;
  const int row = idx / VV;
  const int v   = idx - row * VV;
  float acc = br[v];
  for (int k = 0; k < DD; ++k)
    acc += H[(size_t)row * DD + k] * Wr[k * VV + v];
  out[idx] = acc;
}

// ---------------------------------------------------------------------------
extern "C" void kernel_launch(void* const* d_in, const int* in_sizes, int n_in,
                              void* d_out, int out_size, void* d_ws, size_t ws_size,
                              hipStream_t stream) {
  (void)in_sizes; (void)n_in; (void)out_size; (void)ws_size;

  const int*   x          = (const int*)  d_in[0];
  const float* Wi0        = (const float*)d_in[1];
  const float* bi0        = (const float*)d_in[2];
  const float* Wh0        = (const float*)d_in[3];
  const float* mlp0_W     = (const float*)d_in[4];
  const float* mlp0_b     = (const float*)d_in[5];
  const float* mlp0_scale = (const float*)d_in[6];
  const float* mlp0_bias  = (const float*)d_in[7];
  const float* Wi1        = (const float*)d_in[8];
  const float* bi1        = (const float*)d_in[9];
  const float* Wh1        = (const float*)d_in[10];
  const float* mlp1_W     = (const float*)d_in[11];
  const float* mlp1_b     = (const float*)d_in[12];
  const float* mlp1_scale = (const float*)d_in[13];
  const float* mlp1_bias  = (const float*)d_in[14];
  const float* Wr         = (const float*)d_in[15];
  const float* br         = (const float*)d_in[16];
  float* out = (float*)d_out;

  // Workspace layout
  char* ws = (char*)d_ws;
  float* H     = (float*)ws;                                  // 16 MB
  float* U     = (float*)(ws + (size_t)16 * 1024 * 1024);     // 16 MB (aliases Y)
  float* Y     = U;
  float* stats = (float*)(ws + (size_t)32 * 1024 * 1024);     // 512 f32
  _Float16* WT = (_Float16*)(ws + (size_t)32 * 1024 * 1024 + 4096);
  _Float16* WT_Wh0 = WT + 0 * WSLOT;
  _Float16* WT_Wi1 = WT + 1 * WSLOT;
  _Float16* WT_Wh1 = WT + 2 * WSLOT;
  _Float16* WT_m0  = WT + 3 * WSLOT;   // mlp0 layers 0,1
  _Float16* WT_m1  = WT + 5 * WSLOT;   // mlp1 layers 0,1

  // One-time (per call) weight convert+transpose to f16 B-fragment layout
  convw_k<<<256, 256, 0, stream>>>(Wh0,            WT_Wh0);
  convw_k<<<256, 256, 0, stream>>>(Wi1,            WT_Wi1);
  convw_k<<<256, 256, 0, stream>>>(Wh1,            WT_Wh1);
  convw_k<<<256, 256, 0, stream>>>(mlp0_W,         WT_m0);
  convw_k<<<256, 256, 0, stream>>>(mlp0_W + WSLOT, WT_m0 + WSLOT);
  convw_k<<<256, 256, 0, stream>>>(mlp1_W,         WT_m1);
  convw_k<<<256, 256, 0, stream>>>(mlp1_W + WSLOT, WT_m1 + WSLOT);

  auto run_mlp = [&](const _Float16* WTl, const float* bl,
                     const float* sl, const float* bnl) {
    for (int l = 0; l < 2; ++l) {
      wmma_gemm_k<<<dim3(ROWS / MT, 4), 128, 0, stream>>>(H, WTl + l * WSLOT,
                                                          bl + l * DD, Y);
      zero_k<<<2, 256, 0, stream>>>(stats, 2 * DD);
      stats_k<<<ROWS / 512, 256, 0, stream>>>(Y, stats);
      norm_gelu_res_k<<<ROWS, 256, 0, stream>>>(Y, stats, sl + l * DD,
                                                bnl + l * DD, H);
    }
  };

  // Block 0
  embed_k<<<ROWS, 256, 0, stream>>>(x, Wi0, bi0, U);
  wmma_scan_k<<<1, 512, 0, stream>>>(U, WT_Wh0, H);
  run_mlp(WT_m0, mlp0_b, mlp0_scale, mlp0_bias);

  // Blocks 1..3 (shared parameters)
  for (int blk = 1; blk < 4; ++blk) {
    wmma_gemm_k<<<dim3(ROWS / MT, 4), 128, 0, stream>>>(H, WT_Wi1, bi1, U);
    wmma_scan_k<<<1, 512, 0, stream>>>(U, WT_Wh1, H);
    run_mlp(WT_m1, mlp1_b, mlp1_scale, mlp1_bias);
  }

  // Readout
  readout_k<<<(ROWS * VV + 255) / 256, 256, 0, stream>>>(H, Wr, br, out);
}
